// DynamicHead_57011395887597
// MI455X (gfx1250) — matrix operand order repaired and checked by reading, not compiled, not run on hardware
//
#include <hip/hip_runtime.h>

// ---------------------------------------------------------------------------
// Varying-coefficient MLP on gfx1250 (MI455X).
//   out = sum_s basis[:,s] * (X @ W_s + bias_s)  per layer:
//   X slab resident in LDS as bf16 (converted once), B tiles via
//   GLOBAL_LOAD_ASYNC_TO_LDS_B128, per-s partials in f32 WMMA accumulators
//   seeded with bias_s, basis scaling applied in f32 after each s.
// ---------------------------------------------------------------------------

typedef __attribute__((ext_vector_type(16))) __bf16 v16bf;
typedef __attribute__((ext_vector_type(8)))  float  v8f;

#define BATCH 32768
#define DIN   256
#define NDIM  256
#define SDIM_ 12
#define KDIM  (SDIM_ * 256)   // 3072
#define BM    128
#define BN    128
#define KT    64
#define NT    (KDIM / KT)     // 48
#define LDSK  (KT + 8)        // 72 bf16 -> 144B row stride (16B aligned)
#define LDSX  (DIN + 8)       // 264 bf16 -> 528B row stride (16B aligned)

struct alignas(16) U4 { unsigned int x, y, z, w; };
struct U8pack { U4 lo, hi; };

// Pack two f32 to packed bf16 (round-half-up): 2 adds + 1 v_perm_b32.
__device__ __forceinline__ unsigned int pk_bf16(float a, float b) {
  unsigned int ua = __builtin_bit_cast(unsigned int, a) + 0x8000u;
  unsigned int ub = __builtin_bit_cast(unsigned int, b) + 0x8000u;
  return __builtin_amdgcn_perm(ub, ua, 0x07060302u);
}

__device__ __forceinline__ unsigned short f32_to_bf16(float f) {
  return (unsigned short)((__builtin_bit_cast(unsigned int, f) + 0x8000u) >> 16);
}

__device__ __forceinline__ v16bf pack_frag(U4 lo, U4 hi) {
  U8pack p{lo, hi};
  return __builtin_bit_cast(v16bf, p);
}

__device__ __forceinline__ void wait_async0() {
#if __has_builtin(__builtin_amdgcn_s_wait_asynccnt)
  __builtin_amdgcn_s_wait_asynccnt(0);
#else
  asm volatile("s_wait_asynccnt 0x0" ::: "memory");
#endif
}

// ---------------------------------------------------------------------------
// basis[b, 0:12] = [1, t, t^2, t^3, relu(t - k/9)^3 for k=1..8]
// ---------------------------------------------------------------------------
__global__ __launch_bounds__(256)
void basis_kernel(const float* __restrict__ t, float* __restrict__ basis) {
  int b = blockIdx.x * 256 + threadIdx.x;
  float x  = t[b];
  float x2 = x * x;
  float* o = basis + (size_t)b * SDIM_;
  o[0] = 1.0f; o[1] = x; o[2] = x2; o[3] = x2 * x;
#pragma unroll
  for (int j = 0; j < 8; ++j) {
    float d = fmaxf(x - (float)(j + 1) * (1.0f / 9.0f), 0.0f);
    o[4 + j] = d * d * d;
  }
}

// ---------------------------------------------------------------------------
// Wt[n, k] = bf16(W[k, n]) : one-time convert + transpose (32x32 LDS tiles)
// ---------------------------------------------------------------------------
__global__ __launch_bounds__(256)
void wprep_kernel(const float* __restrict__ W, unsigned short* __restrict__ Wt) {
  __shared__ float tile[32][33];
  const int k0 = blockIdx.x * 32, n0 = blockIdx.y * 32;
  const int tx = threadIdx.x & 31, ty = threadIdx.x >> 5;
#pragma unroll
  for (int i = 0; i < 4; ++i) {
    int k = ty + i * 8;
    tile[k][tx] = W[(size_t)(k0 + k) * NDIM + n0 + tx];
  }
  __syncthreads();
#pragma unroll
  for (int i = 0; i < 4; ++i) {
    int n = ty + i * 8;
    Wt[(size_t)(n0 + n) * KDIM + k0 + tx] = f32_to_bf16(tile[tx][n]);
  }
}

// ---------------------------------------------------------------------------
// grid = (BATCH/BM, NDIM/BN), block = 256 (8 waves, 4x2 over a 128x128 tile)
// ---------------------------------------------------------------------------
__global__ __launch_bounds__(256)
void vc_gemm_kernel(const float* __restrict__ A,             // (BATCH, 256) f32
                    const float* __restrict__ basis,         // (BATCH, 12)  f32
                    const unsigned short* __restrict__ Wt,   // (256, 3072)  bf16
                    const float* __restrict__ bias,          // (12, 256)    f32
                    float* __restrict__ out)                 // (BATCH, 256) f32
{
  __shared__ unsigned short Xs[BM][LDSX];      // resident raw bf16 X slab
  __shared__ unsigned short Bs[2][BN][LDSK];   // [n][k] bf16, ping-pong
  __shared__ float biasS[SDIM_][BN];
  __shared__ float basisT[SDIM_][BM];          // [s][row]

  const int tid  = threadIdx.x;
  const int lane = tid & 31;
  const int wave = tid >> 5;
  const int wm   = wave & 3;      // 4 waves along M: 32 rows each
  const int wn   = wave >> 2;     // 2 waves along N: 64 cols each
  const int lm   = lane & 15;
  const int lh   = lane >> 4;

  const int bm0 = blockIdx.x * BM;
  const int bn0 = blockIdx.y * BN;

  // ---- B async DMA (4 x b128 per thread = 64B of the 128x64 bf16 tile) ----
  const int bRow = tid >> 1;
  const int bCb  = (tid & 1) * 32;
  const unsigned short* bSrcRow = Wt + (size_t)(bn0 + bRow) * KDIM;
  auto async_loadB = [&](int k0, int bb) {
    const unsigned short* g = bSrcRow + k0 + bCb;
    unsigned lds = (unsigned)(size_t)&Bs[bb][bRow][bCb];  // addr[31:0] = LDS addr
#pragma unroll
    for (int v = 0; v < 4; ++v) {
      asm volatile("global_load_async_to_lds_b128 %0, %1, off"
                   :: "v"(lds + 16u * v),
                      "v"((unsigned long long)(size_t)(g + 8 * v))
                   : "memory");
    }
  };
  async_loadB(0, 0);   // overlap DMA of first B tile with the prologue

  // ---- prologue: X slab -> LDS bf16 (once), bias tile, transposed basis ----
  {
    const int row = tid >> 1, c0 = (tid & 1) * 128;   // 128 f32 per thread
    const float4* src = (const float4*)(A + (size_t)(bm0 + row) * DIN + c0);
    U4* dst = (U4*)&Xs[row][c0];
#pragma unroll
    for (int v = 0; v < 16; ++v) {
      float4 f0 = src[2 * v], f1 = src[2 * v + 1];
      U4 u;
      u.x = pk_bf16(f0.x, f0.y);
      u.y = pk_bf16(f0.z, f0.w);
      u.z = pk_bf16(f1.x, f1.y);
      u.w = pk_bf16(f1.z, f1.w);
      dst[v] = u;
    }
  }
  for (int i = tid; i < SDIM_ * BN; i += 256) {
    int s = i >> 7, n = i & 127;
    biasS[s][n] = bias[s * NDIM + bn0 + n];
  }
  for (int i = tid; i < SDIM_ * BM; i += 256) {
    int s = i >> 7, r = i & 127;
    basisT[s][r] = basis[(size_t)(bm0 + r) * SDIM_ + s];
  }

  const v8f zero = {0.f, 0.f, 0.f, 0.f, 0.f, 0.f, 0.f, 0.f};
  v8f facc[2][4];
#pragma unroll
  for (int mf = 0; mf < 2; ++mf)
#pragma unroll
    for (int nf = 0; nf < 4; ++nf) facc[mf][nf] = zero;

  // ---- main loop: s outer, 4 K-tiles of 64 per s, 1 barrier per tile ----
  int buf = 0;
  int g   = 0;                       // flat tile index 0..47
  for (int s = 0; s < SDIM_; ++s) {
    // Seed partials with bias_s[n]: fold produces basis_s*(X@W_s + bias_s).
    v8f pacc[2][4];
#pragma unroll
    for (int nf = 0; nf < 4; ++nf) {
      const float bval = biasS[s][wn * 64 + nf * 16 + lm];
      v8f seed;
#pragma unroll
      for (int r = 0; r < 8; ++r) seed[r] = bval;
      pacc[0][nf] = seed;
      pacc[1][nf] = seed;
    }

    for (int t = 0; t < 4; ++t, ++g) {
      const bool more = (g + 1 < NT);

      wait_async0();        // this wave's B-tile(g) DMA complete
      __syncthreads();      // everyone's DMA (and prologue stores) visible

      // Bs[buf] read-only, Bs[buf^1] write-only until next barrier.
      if (more) async_loadB((g + 1) * KT, buf ^ 1);   // overlaps compute

      const unsigned short (*Bsb)[LDSK] = Bs[buf];
#pragma unroll
      for (int step = 0; step < 2; ++step) {
        const int kb  = step * 32;
        const int xcb = t * 64 + kb;   // column into resident X slab
        v16bf afrag[2];
#pragma unroll
        for (int mf = 0; mf < 2; ++mf) {
          int m = wm * 32 + mf * 16 + lm;
          int kbase = xcb + lh * 8;
          U4 lo = *(const U4*)&Xs[m][kbase];
          U4 hi = *(const U4*)&Xs[m][kbase + 16];
          afrag[mf] = pack_frag(lo, hi);
        }
        v16bf bfrag[4];
#pragma unroll
        for (int nf = 0; nf < 4; ++nf) {
          int n = wn * 64 + nf * 16 + lm;
          int ks = kb + lh * 8;
          U4 lo = *(const U4*)&Bsb[n][ks];
          U4 hi = *(const U4*)&Bsb[n][ks + 16];
          bfrag[nf] = pack_frag(lo, hi);
        }
#pragma unroll
        for (int mf = 0; mf < 2; ++mf)
#pragma unroll
          for (int nf = 0; nf < 4; ++nf)
            pacc[mf][nf] = __builtin_amdgcn_wmma_f32_16x16x32_bf16(
                false, afrag[mf], false, bfrag[nf],
                (short)0, pacc[mf][nf], false, false);
      }
      buf ^= 1;
    }

    // ---- fold partial into final: facc += basis[row, s] * pacc (f32) ----
#pragma unroll
    for (int mf = 0; mf < 2; ++mf) {
      const float* sp = &basisT[s][wm * 32 + mf * 16 + lh * 8];
      float4 s0 = *(const float4*)sp;
      float4 s1 = *(const float4*)(sp + 4);
      const float sc[8] = {s0.x, s0.y, s0.z, s0.w, s1.x, s1.y, s1.z, s1.w};
#pragma unroll
      for (int nf = 0; nf < 4; ++nf)
#pragma unroll
        for (int r = 0; r < 8; ++r)
          facc[mf][nf][r] = fmaf(sc[r], pacc[mf][nf][r], facc[mf][nf][r]);
    }
  }

  // ---- epilogue: relu + store. C layout: M = r + 8*lh, N = lm. ----
#pragma unroll
  for (int mf = 0; mf < 2; ++mf) {
#pragma unroll
    for (int r = 0; r < 8; ++r) {
      const int row = bm0 + wm * 32 + mf * 16 + lh * 8 + r;
#pragma unroll
      for (int nf = 0; nf < 4; ++nf) {
        const int n = wn * 64 + nf * 16 + lm;
        out[(size_t)row * NDIM + bn0 + n] = fmaxf(facc[mf][nf][r], 0.0f);
      }
    }
  }
}

// ---------------------------------------------------------------------------
// Final layer (out width 1): one wave per sample.
// ---------------------------------------------------------------------------
__global__ __launch_bounds__(256)
void final_kernel(const float* __restrict__ h1, const float* __restrict__ basis,
                  const float* __restrict__ W2, const float* __restrict__ b2,
                  float* __restrict__ out) {
  const int wave = threadIdx.x >> 5;
  const int lane = threadIdx.x & 31;
  const int b = blockIdx.x * 8 + wave;

  const float* h  = h1 + (size_t)b * NDIM;
  const float* bp = basis + (size_t)b * SDIM_;
  float bv[SDIM_];
#pragma unroll
  for (int s = 0; s < SDIM_; ++s) bv[s] = bp[s];

  float acc = 0.0f;
#pragma unroll
  for (int j = 0; j < 8; ++j) {
    const int i = lane + j * 32;
    float w = 0.0f;
#pragma unroll
    for (int s = 0; s < SDIM_; ++s) w = fmaf(bv[s], W2[s * NDIM + i], w);
    acc = fmaf(h[i], w, acc);
  }
#pragma unroll
  for (int off = 16; off > 0; off >>= 1) acc += __shfl_down(acc, off, 32);

  if (lane == 0) {
    float bias = 0.0f;
#pragma unroll
    for (int s = 0; s < SDIM_; ++s) bias = fmaf(bv[s], b2[s], bias);
    out[b] = acc + bias;
  }
}

// ---------------------------------------------------------------------------
extern "C" void kernel_launch(void* const* d_in, const int* in_sizes, int n_in,
                              void* d_out, int out_size, void* d_ws, size_t ws_size,
                              hipStream_t stream) {
  const float* treatment = (const float*)d_in[0];
  const float* features  = (const float*)d_in[1];
  const float* W0 = (const float*)d_in[2];
  const float* b0 = (const float*)d_in[3];
  const float* W1 = (const float*)d_in[4];
  const float* b1 = (const float*)d_in[5];
  const float* W2 = (const float*)d_in[6];
  const float* b2 = (const float*)d_in[7];
  float* out = (float*)d_out;

  char* ws = (char*)d_ws;
  float* basis = (float*)ws;                                     // 1.5 MB
  float* h0 = (float*)(ws + (size_t)BATCH * SDIM_ * sizeof(float));
  float* h1 = h0 + (size_t)BATCH * NDIM;                         // 32 MB each
  unsigned short* wt = (unsigned short*)(h1 + (size_t)BATCH * NDIM); // 1.5 MB

  basis_kernel<<<BATCH / 256, 256, 0, stream>>>(treatment, basis);

  dim3 gprep(KDIM / 32, NDIM / 32);   // (96, 8)
  dim3 ggemm(BATCH / BM, NDIM / BN);  // (256, 2)

  wprep_kernel<<<gprep, 256, 0, stream>>>(W0, wt);
  vc_gemm_kernel<<<ggemm, 256, 0, stream>>>(features, basis, wt, b0, h0);

  wprep_kernel<<<gprep, 256, 0, stream>>>(W1, wt);
  vc_gemm_kernel<<<ggemm, 256, 0, stream>>>(h0, basis, wt, b1, h1);

  final_kernel<<<BATCH / 8, 256, 0, stream>>>(h1, basis, W2, b2, out);
}
